// KascadeAnchorAttention_59657095741804
// MI455X (gfx1250) — matrix operand order repaired and checked by reading, not compiled, not run on hardware
//
#include <hip/hip_runtime.h>
#include <hip/hip_bf16.h>

// ---------------------------------------------------------------------------
// Causal MHA for MI455X (gfx1250): bf16 WMMA, fp32 accumulate, TDM-staged
// LDS panels for the GEMMs. B=2, S=2048, H=16, Dh=64, Dm=1024.
// ---------------------------------------------------------------------------

typedef __bf16 bf16_t;
typedef __attribute__((ext_vector_type(16))) __bf16 v16bf;
typedef __attribute__((ext_vector_type(8)))  float  v8f;

typedef unsigned int u32x4 __attribute__((ext_vector_type(4)));
typedef int          i32x8 __attribute__((ext_vector_type(8)));
typedef int          i32x4 __attribute__((ext_vector_type(4)));

#define S_LEN   2048
#define HEADS   16
#define HDIM    64
#define DMODEL  1024
#define BATCH   2

#if defined(__AMDGCN__) && __has_builtin(__builtin_amdgcn_tensor_load_to_lds) && \
    __has_builtin(__builtin_amdgcn_s_wait_tensorcnt)
#define HAVE_TDM 1
#else
#define HAVE_TDM 0
#endif

static __device__ __forceinline__ v8f wmma_bf16(v16bf a, v16bf b, v8f c) {
    // (neg_a, A, neg_b, B, c_mod, C, reuse_a, reuse_b)
    return __builtin_amdgcn_wmma_f32_16x16x32_bf16(false, a, false, b,
                                                   (short)0, c, false, false);
}

template <typename T> struct dsz_code;
template <> struct dsz_code<float>  { static constexpr unsigned v = 2; }; // 4B
template <> struct dsz_code<bf16_t> { static constexpr unsigned v = 1; }; // 2B

#if HAVE_TDM
// Issue a 2D TDM tile load (Global -> LDS). Descriptor layout per CDNA5 ISA
// ch.8: group0 = {flags, lds_addr, global_addr, type=2}; group1 packs
// data_size, tensor dims, tile dims, dim0 stride. All values wave-uniform.
static __device__ __forceinline__ void tdm_load_2d(
    unsigned lds_off, const void* gaddr,
    unsigned tensor_d0, unsigned tensor_d1, unsigned row_stride,
    unsigned tile_d0, unsigned tile_d1, unsigned data_size)
{
    unsigned long long ga = (unsigned long long)gaddr;
    u32x4 g0;
    g0[0] = 1u;                                     // count=1, user descriptor
    g0[1] = lds_off;                                // lds_addr (bytes)
    g0[2] = (unsigned)(ga & 0xffffffffull);         // global_addr[31:0]
    g0[3] = (unsigned)((ga >> 32) & 0x1ffffffull)   // global_addr[56:32]
          | (2u << 30);                             // type = 2 ("image")
    i32x8 g1;
    g1[0] = (int)(data_size << 16);                 // wg_mask=0, data_size
    g1[1] = (int)((tensor_d0 & 0xffffu) << 16);     // tensor_dim0[15:0]
    g1[2] = (int)((tensor_d0 >> 16) | ((tensor_d1 & 0xffffu) << 16));
    g1[3] = (int)((tensor_d1 >> 16) | (tile_d0 << 16));   // tile_dim0
    g1[4] = (int)(tile_d1 & 0xffffu);               // tile_dim1 (tile_dim2=0)
    g1[5] = (int)row_stride;                        // tensor_dim0_stride[31:0]
    g1[6] = 0;                                      // stride[47:32], dim1_stride lo
    g1[7] = 0;
    i32x4 z4 = {0, 0, 0, 0};
#if __clang_major__ >= 23
    i32x8 z8 = {0, 0, 0, 0, 0, 0, 0, 0};
    __builtin_amdgcn_tensor_load_to_lds(g0, g1, z4, z4, z8, 0);
#else
    __builtin_amdgcn_tensor_load_to_lds(g0, g1, z4, z4, 0);
#endif
}

static __device__ __forceinline__ unsigned lds_off_u32(const void* p) {
    // Generic LDS pointer: low 32 bits are the LDS byte offset (aperture map).
    return (unsigned)(unsigned long long)(size_t)p;
}
#endif  // HAVE_TDM

// ---------------------------------------------------------------------------
// Tiled GEMM: Out(OT, MxN) = A(AT, MxK) @ W(f32, KxN).
// Block = 256 threads (8 waves) -> 128x64 output tile, LDS-staged panels.
// Each wave owns a 32x32 register block = 2x2 WMMA fragments.
// ---------------------------------------------------------------------------
template <typename AT, typename OT>
__global__ __launch_bounds__(256)
void tile_gemm(const AT* __restrict__ A, const float* __restrict__ W,
               OT* __restrict__ Out, int M, int N, int K)
{
    constexpr int MT = 128, NT = 64, KB = 32;
    __shared__ AT    shA[MT * KB];   // 128x32 panel of A (row-major)
    __shared__ float shB[KB * NT];   // 32x64 panel of W (row-major)

    const int tid  = threadIdx.x;
    const int lane = tid & 31;
    const int wave = tid >> 5;       // 0..7
    const int wm   = wave & 3;       // 4 waves along M
    const int wn   = wave >> 2;      // 2 waves along N
    const int half = lane >> 4;
    const int lm   = lane & 15;
    const int m0   = blockIdx.x * MT;
    const int n0   = blockIdx.y * NT;

    v8f acc[2][2] = {};

    for (int k0 = 0; k0 < K; k0 += KB) {
#if HAVE_TDM
        if (tid < 32) {   // wave 0 drives the Tensor Data Mover
            tdm_load_2d(lds_off_u32(shA), A + (size_t)m0 * K + k0,
                        (unsigned)K, (unsigned)M, (unsigned)K,
                        KB, MT, dsz_code<AT>::v);
            tdm_load_2d(lds_off_u32(shB), W + (size_t)k0 * N + n0,
                        (unsigned)N, (unsigned)K, (unsigned)N,
                        NT, KB, dsz_code<float>::v);
            __builtin_amdgcn_s_wait_tensorcnt(0);
        }
#else
        #pragma unroll 4
        for (int i = tid; i < MT * KB; i += 256) {
            const int r = i >> 5, c = i & (KB - 1);
            shA[i] = A[(size_t)(m0 + r) * K + k0 + c];
        }
        #pragma unroll 4
        for (int i = tid; i < KB * NT; i += 256) {
            const int r = i >> 6, c = i & (NT - 1);
            shB[i] = W[(size_t)(k0 + r) * N + n0 + c];
        }
#endif
        __syncthreads();

        // A fragments: lane = row, contiguous K order k = 16*half + s, same as B.
        v16bf af[2];
        #pragma unroll
        for (int mi = 0; mi < 2; ++mi) {
            const AT* ap = shA + (wm * 32 + mi * 16 + lm) * KB + 16 * half;
            #pragma unroll
            for (int j = 0; j < 8; ++j) {
                af[mi][j]     = (bf16_t)ap[j];
                af[mi][8 + j] = (bf16_t)ap[8 + j];
            }
        }
        // B fragments: lane = column, K = 16*half..+15, same order as A.
        v16bf bfr[2];
        #pragma unroll
        for (int ni = 0; ni < 2; ++ni) {
            const float* bp = shB + (16 * half) * NT + wn * 32 + ni * 16 + lm;
            #pragma unroll
            for (int i = 0; i < 16; ++i) bfr[ni][i] = (bf16_t)bp[i * NT];
        }
        #pragma unroll
        for (int mi = 0; mi < 2; ++mi)
            #pragma unroll
            for (int ni = 0; ni < 2; ++ni)
                acc[mi][ni] = wmma_bf16(af[mi], bfr[ni], acc[mi][ni]);

        __syncthreads();
    }

    // C layout: lane holds (m = r + 8*half, n = lm).
    #pragma unroll
    for (int mi = 0; mi < 2; ++mi)
        #pragma unroll
        for (int ni = 0; ni < 2; ++ni)
            #pragma unroll
            for (int r = 0; r < 8; ++r)
                Out[(size_t)(m0 + wm * 32 + mi * 16 + r + 8 * half) * N
                    + n0 + wn * 32 + ni * 16 + lm] = (OT)acc[mi][ni][r];
}

// ---------------------------------------------------------------------------
// Flash attention, causal. One wave handles 16 queries of one (batch, head).
// Q/K/V stored (B*S, DMODEL) bf16; output O same layout (bf16).
// ---------------------------------------------------------------------------
__global__ __launch_bounds__(32)
void flash_attn(const bf16_t* __restrict__ Q, const bf16_t* __restrict__ Kc,
                const bf16_t* __restrict__ V, bf16_t* __restrict__ O)
{
    __shared__ float pbuf[16][33];   // 16 queries x 32 keys (padded)

    const int lane = threadIdx.x & 31;
    const int half = lane >> 4;
    const int lm   = lane & 15;
    const int q0   = blockIdx.x * 16;
    const int h    = blockIdx.y;
    const int b    = blockIdx.z;
    const size_t base = ((size_t)b * S_LEN) * DMODEL + (size_t)h * HDIM;

    // Q tile in A-layout for the two 32-wide d-chunks of HDIM=64.
    v16bf aq[2];
    #pragma unroll
    for (int t = 0; t < 2; ++t) {
        const bf16_t* qp = Q + base + (size_t)(q0 + lm) * DMODEL + 32 * t + 16 * half;
        #pragma unroll
        for (int j = 0; j < 8; ++j) {
            aq[t][j]     = qp[j];
            aq[t][8 + j] = qp[8 + j];
        }
    }

    float mrun[8], lrun[8];
    v8f   oacc[4];
    #pragma unroll
    for (int r = 0; r < 8; ++r) { mrun[r] = -1.0e30f; lrun[r] = 0.0f; }
    #pragma unroll
    for (int t = 0; t < 4; ++t) oacc[t] = v8f{};

    const int kmax = q0 + 15;                  // causal: last key needed
    for (int k0 = 0; k0 <= kmax; k0 += 32) {
        // Warm L2/WGP$ for the V rows of this key block (global_prefetch_b8).
        __builtin_prefetch(V + base + (size_t)(k0 + lane) * DMODEL, 0, 1);

        // ---- logits: 16 queries x 32 keys, contraction over HDIM=64 ----
        v8f lg[2] = {};
        #pragma unroll
        for (int kt = 0; kt < 2; ++kt) {
            #pragma unroll
            for (int t = 0; t < 2; ++t) {
                const bf16_t* kp = Kc + base +
                    (size_t)(k0 + 16 * kt + lm) * DMODEL + 32 * t + 16 * half;
                v16bf bk;
                #pragma unroll
                for (int i = 0; i < 16; ++i) bk[i] = kp[i];
                lg[kt] = wmma_bf16(aq[t], bk, lg[kt]);
            }
        }

        // ---- online softmax (rows live across lanes 0-15 / 16-31) ----
        #pragma unroll
        for (int r = 0; r < 8; ++r) {
            const int qrow = q0 + r + 8 * half;
            float l0 = lg[0][r] * 0.125f;          // 1/sqrt(64)
            float l1 = lg[1][r] * 0.125f;
            if (k0 + lm > qrow)      l0 = -1.0e30f;
            if (k0 + 16 + lm > qrow) l1 = -1.0e30f;
            float mx = fmaxf(l0, l1);
            #pragma unroll
            for (int d = 8; d >= 1; d >>= 1) mx = fmaxf(mx, __shfl_xor(mx, d, 32));
            const float mnew = fmaxf(mrun[r], mx);
            const float scl  = __expf(mrun[r] - mnew);
            const float p0   = __expf(l0 - mnew);
            const float p1   = __expf(l1 - mnew);
            float rs = p0 + p1;
            #pragma unroll
            for (int d = 8; d >= 1; d >>= 1) rs += __shfl_xor(rs, d, 32);
            lrun[r] = lrun[r] * scl + rs;
            mrun[r] = mnew;
            #pragma unroll
            for (int t = 0; t < 4; ++t) oacc[t][r] *= scl;
            pbuf[r + 8 * half][lm]      = p0;
            pbuf[r + 8 * half][16 + lm] = p1;
        }
        __syncthreads();

        // ---- transpose P through LDS into WMMA A-layout (bf16) ----
        v16bf pa;
        #pragma unroll
        for (int v = 0; v < 8; ++v) {
            const int kk = 16 * half + 2 * v;
            pa[2 * v]     = (bf16_t)pbuf[lm][kk];
            pa[2 * v + 1] = (bf16_t)pbuf[lm][kk + 1];
        }
        __syncthreads();

        // ---- O += P @ V  (4 output d-chunks of 16) ----
        #pragma unroll
        for (int nt = 0; nt < 4; ++nt) {
            const bf16_t* vp = V + base +
                (size_t)(k0 + 16 * half) * DMODEL + nt * 16 + lm;
            v16bf bv;
            #pragma unroll
            for (int i = 0; i < 16; ++i) bv[i] = vp[(size_t)i * DMODEL];
            oacc[nt] = wmma_bf16(pa, bv, oacc[nt]);
        }
    }

    // ---- finalize: divide by row sums, store bf16 ----
    #pragma unroll
    for (int r = 0; r < 8; ++r) {
        const float inv = 1.0f / lrun[r];
        #pragma unroll
        for (int nt = 0; nt < 4; ++nt) {
            O[base + (size_t)(q0 + r + 8 * half) * DMODEL + nt * 16 + lm] =
                (bf16_t)(oacc[nt][r] * inv);
        }
    }
}

// ---------------------------------------------------------------------------
extern "C" void kernel_launch(void* const* d_in, const int* in_sizes, int n_in,
                              void* d_out, int out_size, void* d_ws, size_t ws_size,
                              hipStream_t stream) {
    const float* x  = (const float*)d_in[0];
    const float* wq = (const float*)d_in[1];
    const float* wk = (const float*)d_in[2];
    const float* wv = (const float*)d_in[3];
    const float* wo = (const float*)d_in[4];
    float* out = (float*)d_out;

    const int M = BATCH * S_LEN;   // 4096 token rows
    const int N = DMODEL;          // 1024
    const int K = DMODEL;          // 1024

    // Workspace: Q, K, V, attn-out, each M*N bf16 (8 MB) -> 32 MB total.
    bf16_t* qws = (bf16_t*)d_ws;
    bf16_t* kws = qws + (size_t)M * N;
    bf16_t* vws = kws + (size_t)M * N;
    bf16_t* aws = vws + (size_t)M * N;

    dim3 gblk(256);
    dim3 ggrid(M / 128, N / 64);

    tile_gemm<float, bf16_t><<<ggrid, gblk, 0, stream>>>(x, wq, qws, M, N, K);
    tile_gemm<float, bf16_t><<<ggrid, gblk, 0, stream>>>(x, wk, kws, M, N, K);
    tile_gemm<float, bf16_t><<<ggrid, gblk, 0, stream>>>(x, wv, vws, M, N, K);

    dim3 attn_grid(S_LEN / 16, HEADS, BATCH);
    flash_attn<<<attn_grid, dim3(32), 0, stream>>>(qws, kws, vws, aws);

    tile_gemm<bf16_t, float><<<ggrid, gblk, 0, stream>>>(aws, wo, out, M, N, K);
}
